// CVAE_1563368096116
// MI455X (gfx1250) — matrix operand order repaired and checked by reading, not compiled, not run on hardware
//
#include <hip/hip_runtime.h>
#include <hip/hip_bf16.h>
#include <math.h>
#include <stdint.h>

// ---------------- problem constants ----------------
constexpr int BB  = 4096;   // batch
constexpr int LL  = 50;     // history length
constexpr int TT  = 72;     // future length
constexpr int XDI = 6;
constexpr int YDI = 2;
constexpr int ZDI = 16;
constexpr int HH  = 128;
constexpr int WID = 256;
constexpr int KK  = 8;
constexpr int RR  = (KK + 1) * BB;   // 36864 batched decode rows
constexpr float C_LOG2PI = 1.8378770664093453f;

typedef __attribute__((ext_vector_type(16))) __bf16 v16bf;
typedef __attribute__((ext_vector_type(8)))  float  v8f;

// ---------------- helpers ----------------
__device__ __forceinline__ uint32_t bfbits(float f) {
    union { float f; uint32_t u; } c; c.f = f;
    uint32_t u = c.u;
    u += 0x7fffu + ((u >> 16) & 1u);            // round-to-nearest-even
    return u >> 16;
}
__device__ __forceinline__ uint32_t pk2(float lo, float hi) {
    return bfbits(lo) | (bfbits(hi) << 16);
}
__device__ __forceinline__ v16bf mkfrag(uint4 lo, uint4 hi) {
    v16bf r;
    __builtin_memcpy(&r, &lo, 16);
    __builtin_memcpy(((char*)&r) + 16, &hi, 16);
    return r;
}
__device__ __forceinline__ unsigned int ihash(unsigned int x) {
    x ^= x >> 16; x *= 0x7feb352dU;
    x ^= x >> 15; x *= 0x846ca68bU;
    x ^= x >> 16; return x;
}
__device__ __forceinline__ float postop(float x, const float* bias, int n, int op) {
    if (op != 0) x += bias[n];
    if (op == 2) x = fmaxf(x, 0.0f);                                        // ReLU
    else if (op == 3) x = 0.5f * x * (1.0f + erff(x * 0.70710678118654752f)); // exact GELU
    else if (op == 4) x = tanhf(x);
    return x;
}

// ---------------- WMMA bf16 GEMM: C[M,N] = op(A[M,K] @ W[N,K]^T + bias) ----------------
// 128x64 C tile per workgroup (8 waves, each wave 32x32 = 4 WMMAs per K-tile).
// LDS holds bf16 pairs packed in uint32, laid out in WMMA fragment order:
//   Asp[m][p]  : pair p = A[m][2p],A[m][2p+1]      (row stride 20 u32 = 80B, 16B aligned)
//   Bsp[n][p]  : pair p = W[n][2p],W[n][2p+1]      (== B[2p][n],B[2p+1][n], per-VGPR packing)
__global__ __launch_bounds__(256)
void k_gemm(const float* __restrict__ A, int lda,
            const float* __restrict__ W,   // [N][K] row-major weight
            const float* __restrict__ bias,
            float* __restrict__ C, int ldc,
            int M, int N, int K, int op)
{
    constexpr int APITCH = 20;   // u32 pitch (16 pairs + pad), 80B = 16B-multiple
    __shared__ uint32_t Asp[128 * APITCH];
    __shared__ uint32_t Bsp[64 * APITCH];

    const int tid  = threadIdx.x;
    const int lane = tid & 31;
    const int wv   = tid >> 5;
    const int hlf  = lane >> 4;        // 0: lanes 0-15, 1: lanes 16-31
    const int mr   = lane & 15;
    const int bm   = blockIdx.x * 128;
    const int bn   = blockIdx.y * 64;
    const int wm   = (wv >> 1) * 32;   // wave row offset (4 M-slots)
    const int wn   = (wv & 1) * 32;    // wave col offset (2 N-slots)

    v8f c00 = {}, c01 = {}, c10 = {}, c11 = {};

    const int KT = (K + 31) >> 5;
    for (int kt = 0; kt < KT; ++kt) {
        const int k0 = kt << 5;
        const bool interior = (bm + 128 <= M) && (bn + 64 <= N) && (k0 + 32 <= K);

        if (interior) {
            // ---- A: 128 rows x 16 pairs; thread -> 1 half-row (8 pairs) ----
            {
                int m  = tid >> 1;
                int p0 = (tid & 1) * 8;                 // pair start (k offset = 2*p0)
                const float* ap = A + (size_t)(bm + m) * lda + k0 + 2 * p0;
                float4 f0 = *(const float4*)(ap);
                float4 f1 = *(const float4*)(ap + 4);
                float4 f2 = *(const float4*)(ap + 8);
                float4 f3 = *(const float4*)(ap + 12);
                uint32_t* dst = &Asp[m * APITCH + p0];
                uint4 q0 = { pk2(f0.x, f0.y), pk2(f0.z, f0.w), pk2(f1.x, f1.y), pk2(f1.z, f1.w) };
                uint4 q1 = { pk2(f2.x, f2.y), pk2(f2.z, f2.w), pk2(f3.x, f3.y), pk2(f3.z, f3.w) };
                *(uint4*)(dst)     = q0;
                *(uint4*)(dst + 4) = q1;
            }
            // ---- B: 64 cols x 16 pairs; thread -> 4 pairs (8 consecutive k of one col) ----
            {
                int n  = tid >> 2;
                int ch = tid & 3;                       // pair chunk
                const float* wp = W + (size_t)(bn + n) * K + k0 + 8 * ch;
                float4 g0 = *(const float4*)(wp);
                float4 g1 = *(const float4*)(wp + 4);
                uint4 q = { pk2(g0.x, g0.y), pk2(g0.z, g0.w), pk2(g1.x, g1.y), pk2(g1.z, g1.w) };
                *(uint4*)&Bsp[n * APITCH + 4 * ch] = q;
            }
        } else {
            // ---- guarded scalar path (edge tiles only) ----
            #pragma unroll
            for (int i = 0; i < 8; ++i) {               // A: 2048 pairs
                int idx = tid * 8 + i;
                int m = idx >> 4, p = idx & 15;
                int gm = bm + m, gk = k0 + 2 * p;
                float lo = (gm < M && gk     < K) ? A[(long)gm * lda + gk]     : 0.0f;
                float hi = (gm < M && gk + 1 < K) ? A[(long)gm * lda + gk + 1] : 0.0f;
                Asp[m * APITCH + p] = pk2(lo, hi);
            }
            #pragma unroll
            for (int i = 0; i < 4; ++i) {               // B: 1024 pairs
                int idx = tid * 4 + i;
                int n = idx >> 4, p = idx & 15;
                int gn = bn + n, gk = k0 + 2 * p;
                float lo = (gn < N && gk     < K) ? W[(long)gn * K + gk]     : 0.0f;
                float hi = (gn < N && gk + 1 < K) ? W[(long)gn * K + gk + 1] : 0.0f;
                Bsp[n * APITCH + p] = pk2(lo, hi);
            }
        }
        __syncthreads();

        if (kt + 1 < KT) {   // speculative prefetch of next tiles -> global_prefetch_b8
            __builtin_prefetch(A + (size_t)(bm + (tid >> 1)) * lda + (k0 + 32), 0, 1);
            __builtin_prefetch(W + (size_t)(bn + (tid >> 2)) * K + (k0 + 32), 0, 1);
        }

        // ---- fragments: contiguous 16B LDS chunks -> ds_load_b128 ----
        // A frag pairs: VGPR i<4 -> hlf*4+i ; i>=4 -> 8+hlf*4+(i-4)
        v16bf a0 = mkfrag(*(const uint4*)&Asp[(wm + mr) * APITCH + hlf * 4],
                          *(const uint4*)&Asp[(wm + mr) * APITCH + 8 + hlf * 4]);
        v16bf a1 = mkfrag(*(const uint4*)&Asp[(wm + 16 + mr) * APITCH + hlf * 4],
                          *(const uint4*)&Asp[(wm + 16 + mr) * APITCH + 8 + hlf * 4]);
        // B frag pairs: VGPR i<4 -> hlf*4+i ; i>=4 -> 8+hlf*4+(i-4)
        v16bf b0 = mkfrag(*(const uint4*)&Bsp[(wn + mr) * APITCH + hlf * 4],
                          *(const uint4*)&Bsp[(wn + mr) * APITCH + 8 + hlf * 4]);
        v16bf b1 = mkfrag(*(const uint4*)&Bsp[(wn + 16 + mr) * APITCH + hlf * 4],
                          *(const uint4*)&Bsp[(wn + 16 + mr) * APITCH + 8 + hlf * 4]);

        c00 = __builtin_amdgcn_wmma_f32_16x16x32_bf16(false, a0, false, b0, (short)0, c00, false, false);
        c01 = __builtin_amdgcn_wmma_f32_16x16x32_bf16(false, a0, false, b1, (short)0, c01, false, false);
        c10 = __builtin_amdgcn_wmma_f32_16x16x32_bf16(false, a1, false, b0, (short)0, c10, false, false);
        c11 = __builtin_amdgcn_wmma_f32_16x16x32_bf16(false, a1, false, b1, (short)0, c11, false, false);
        __syncthreads();
    }

    // ---- store: C VGPR r -> M = r + 8*hlf, N = lane&15 ----
    #pragma unroll
    for (int r = 0; r < 8; ++r) {
        int m0 = bm + wm + r + 8 * hlf;
        int m1 = m0 + 16;
        int n0 = bn + wn + mr;
        int n1 = n0 + 16;
        if (m0 < M) {
            if (n0 < N) C[(long)m0 * ldc + n0] = postop(c00[r], bias, n0, op);
            if (n1 < N) C[(long)m0 * ldc + n1] = postop(c01[r], bias, n1, op);
        }
        if (m1 < M) {
            if (n0 < N) C[(long)m1 * ldc + n0] = postop(c10[r], bias, n0, op);
            if (n1 < N) C[(long)m1 * ldc + n1] = postop(c11[r], bias, n1, op);
        }
    }
}

// ---------------- GRU cell: h = (1-z)*n + z*h ----------------
__global__ void k_gru(float* __restrict__ h, const float* __restrict__ gi,
                      const float* __restrict__ gh, int rows)
{
    long i = (long)blockIdx.x * blockDim.x + threadIdx.x;
    long n = (long)rows * HH;
    if (i >= n) return;
    long b = i >> 7; int g = (int)(i & 127);
    const float* gir = gi + b * (3 * HH);
    const float* ghr = gh + b * (3 * HH);
    float ir = gir[g], iz = gir[HH + g], in = gir[2 * HH + g];
    float hr = ghr[g], hz = ghr[HH + g], hn = ghr[2 * HH + g];
    float r  = 1.0f / (1.0f + expf(-(ir + hr)));
    float z  = 1.0f / (1.0f + expf(-(iz + hz)));
    float nn = tanhf(in + r * hn);
    float hp = h[i];
    h[i] = (1.0f - z) * nn + z * hp;
}

// ---------------- residual LayerNorm over 256 cols, in-place into x ----------------
__global__ __launch_bounds__(256)
void k_ln_res(float* __restrict__ x, const float* __restrict__ y,
              const float* __restrict__ g, const float* __restrict__ be, int rows)
{
    __shared__ float red[256];
    long row = blockIdx.x;
    if (row >= rows) return;
    int c = threadIdx.x;
    float v = x[row * WID + c] + y[row * WID + c];
    red[c] = v; __syncthreads();
    for (int s = 128; s > 0; s >>= 1) { if (c < s) red[c] += red[c + s]; __syncthreads(); }
    float mean = red[0] * (1.0f / WID);
    __syncthreads();
    float d = v - mean;
    red[c] = d * d; __syncthreads();
    for (int s = 128; s > 0; s >>= 1) { if (c < s) red[c] += red[c + s]; __syncthreads(); }
    float var = red[0] * (1.0f / WID);
    x[row * WID + c] = d * rsqrtf(var + 1e-5f) * g[c] + be[c];
}

// ---------------- small utility kernels ----------------
__global__ void k_zero(float* __restrict__ p, long n) {
    long i = (long)blockIdx.x * blockDim.x + threadIdx.x;
    if (i < n) p[i] = 0.0f;
}

__global__ void k_concat2(float* __restrict__ dst, const float* __restrict__ a, int ca,
                          const float* __restrict__ b, int cb, int rows)
{
    long i = (long)blockIdx.x * blockDim.x + threadIdx.x;
    int cols = ca + cb;
    if (i >= (long)rows * cols) return;
    long r = i / cols; int c = (int)(i % cols);
    dst[i] = (c < ca) ? a[r * ca + c] : b[r * cb + (c - ca)];
}

// z (s==0 posterior, s>=1 prior) with deterministic Box-Muller noise
__global__ void k_sample_z(const float* __restrict__ po_out, const float* __restrict__ pr_out,
                           float* __restrict__ Z)
{
    long i = (long)blockIdx.x * blockDim.x + threadIdx.x;
    if (i >= (long)RR * ZDI) return;
    int r = (int)(i / ZDI), d = (int)(i % ZDI);
    int s = r / BB, b = r % BB;
    float mu, lv;
    if (s == 0) { mu = po_out[b * 32 + d]; lv = po_out[b * 32 + 16 + d]; }
    else        { mu = pr_out[b * 32 + d]; lv = pr_out[b * 32 + 16 + d]; }
    unsigned int seed = (unsigned int)(((s * BB + b) * ZDI + d));
    float u1 = (ihash(seed * 2u + 1u) + 0.5f) * 2.3283064365386963e-10f;
    float u2 = (ihash(seed * 2u + 2u) + 0.5f) * 2.3283064365386963e-10f;
    float nrm = sqrtf(-2.0f * logf(u1)) * cosf(6.283185307179586f * u2);
    Z[(long)r * ZDI + d] = mu + nrm * (expf(0.5f * lv) * 1.5f);
}

__global__ void k_build_cond(float* __restrict__ cond, const float* __restrict__ henc,
                             const float* __restrict__ Z)
{
    long i = (long)blockIdx.x * blockDim.x + threadIdx.x;
    const int cols = HH + ZDI;     // 144
    if (i >= (long)RR * cols) return;
    int r = (int)(i / cols), c = (int)(i % cols);
    int b = r % BB;
    cond[i] = (c < HH) ? henc[(long)b * HH + c] : Z[(long)r * ZDI + (c - HH)];
}

__global__ void k_build_vz(float* __restrict__ vz, const float* __restrict__ v,
                           const float* __restrict__ Z)
{
    long i = (long)blockIdx.x * blockDim.x + threadIdx.x;
    const int cols = YDI + ZDI;    // 18
    if (i >= (long)RR * cols) return;
    int r = (int)(i / cols), c = (int)(i % cols);
    vz[i] = (c < YDI) ? v[(long)r * YDI + c] : Z[(long)r * ZDI + (c - YDI)];
}

__global__ void k_init_pos_v(float* __restrict__ pos, float* __restrict__ v,
                             const float* __restrict__ x_hist, const float* __restrict__ v0)
{
    long i = (long)blockIdx.x * blockDim.x + threadIdx.x;
    if (i >= (long)RR * YDI) return;
    int r = (int)(i / YDI), j = (int)(i % YDI);
    int b = r % BB;
    pos[i] = x_hist[(long)b * LL * XDI + (LL - 1) * XDI + j];   // last_pos = x_hist[:,-1,:2]
    v[i]   = v0[j];
}

// ---------------- per-step head finish: mu/lv dots, softplus, pos integrate ----------------
__global__ void k_heads(const float* __restrict__ x,
                        const float* __restrict__ muw, const float* __restrict__ mub,
                        const float* __restrict__ lvw, const float* __restrict__ lvb,
                        float* __restrict__ pos, float* __restrict__ v,
                        float* __restrict__ mu_main, float* __restrict__ lv_main,
                        float* __restrict__ trajs, int t)
{
    int r = blockIdx.x * blockDim.x + threadIdx.x;
    if (r >= RR) return;
    const float* xr = x + (long)r * WID;
    int s = r / BB, b = r % BB;
    #pragma unroll
    for (int j = 0; j < YDI; ++j) {
        float am = 0.0f, al = 0.0f;
        for (int c = 0; c < WID; c += 4) {
            float4 xv = *(const float4*)(xr + c);
            float4 mw = *(const float4*)(muw + j * WID + c);
            float4 lw = *(const float4*)(lvw + j * WID + c);
            am += xv.x * mw.x + xv.y * mw.y + xv.z * mw.z + xv.w * mw.w;
            al += xv.x * lw.x + xv.y * lw.y + xv.z * lw.z + xv.w * lw.w;
        }
        float delta = am + mub[j];
        float rl = fminf(fmaxf(al + lvb[j], -10.0f), 10.0f);
        float sp = (rl > 20.0f) ? rl : log1pf(expf(rl));
        float var = fminf(sp + 0.01f, 10.0f);
        float lvout = logf(var);
        float p = pos[(long)r * YDI + j] + delta;
        pos[(long)r * YDI + j] = p;
        v[(long)r * YDI + j] = delta;                 // carry stop_gradient(delta)
        if (s == 0) {
            mu_main[((long)b * TT + t) * YDI + j] = p;
            lv_main[((long)b * TT + t) * YDI + j] = lvout;
        } else {
            trajs[(((long)(s - 1) * BB + b) * TT + t) * YDI + j] = p;
        }
    }
}

// ---------------- losses ----------------
__global__ void k_nll_mse_calib(const float* __restrict__ mu, const float* __restrict__ lv,
                                const float* __restrict__ y, float* __restrict__ sums)
{
    int i = blockIdx.x * blockDim.x + threadIdx.x;
    if (i >= BB * TT) return;
    float snll = 0.0f, smse = 0.0f, sd2 = 0.0f;
    const float lmin = logf(0.01f), lmax = logf(100.0f);
    #pragma unroll
    for (int j = 0; j < YDI; ++j) {
        long idx = (long)i * YDI + j;
        float m = mu[idx], l = lv[idx], yv = y[idx];
        float e = yv - m;
        float lvc = fminf(fmaxf(l, lmin), lmax);
        snll += 0.5f * (C_LOG2PI + lvc + e * e / expf(lvc));
        smse += e * e;
        float var = fmaxf(expf(l), 1e-12f);
        sd2 += e * e / (var + 0.001f);
    }
    atomicAdd(&sums[0], snll);
    atomicAdd(&sums[1], smse);
    atomicAdd(&sums[2], sd2);
}

__global__ void k_kl(const float* __restrict__ po_out, const float* __restrict__ pr_out,
                     float* __restrict__ sums)
{
    int b = blockIdx.x * blockDim.x + threadIdx.x;
    if (b >= BB) return;
    const float lmin = logf(1e-5f), lmax = logf(100.0f);
    float acc = 0.0f;
    for (int d = 0; d < ZDI; ++d) {
        float mq = po_out[b * 32 + d], lq = po_out[b * 32 + 16 + d];
        float mp = pr_out[b * 32 + d], lp = pr_out[b * 32 + 16 + d];
        lq = fminf(fmaxf(lq, lmin), lmax);
        lp = fminf(fmaxf(lp, lmin), lmax);
        float dm = mq - mp;
        acc += 0.5f * (lp - lq + (expf(lq) + dm * dm) / expf(lp) - 1.0f);
    }
    atomicAdd(&sums[3], acc);
}

__global__ void k_cov(const float* __restrict__ trajs, const float* __restrict__ y,
                      float* __restrict__ sums)
{
    int i = blockIdx.x * blockDim.x + threadIdx.x;
    if (i >= BB * TT) return;
    int b = i / TT, t = i % TT;
    float mb0 = 0.0f, mb1 = 0.0f;
    #pragma unroll
    for (int k = 0; k < KK; ++k) {
        long base = (((long)k * BB + b) * TT + t) * YDI;
        mb0 += trajs[base]; mb1 += trajs[base + 1];
    }
    mb0 *= (1.0f / KK); mb1 *= (1.0f / KK);
    float cxx = 0.0f, cxy = 0.0f, cyy = 0.0f;
    #pragma unroll
    for (int k = 0; k < KK; ++k) {
        long base = (((long)k * BB + b) * TT + t) * YDI;
        float d0 = trajs[base] - mb0, d1 = trajs[base + 1] - mb1;
        cxx += d0 * d0; cxy += d0 * d1; cyy += d1 * d1;
    }
    const float inv = 1.0f / (KK - 1);
    cxx = cxx * inv + 1e-6f; cyy = cyy * inv + 1e-6f; cxy *= inv;
    long yb = ((long)b * TT + t) * YDI;
    float df0 = y[yb] - mb0, df1 = y[yb + 1] - mb1;
    float det = cxx * cyy - cxy * cxy;
    float s0 = (cyy * df0 - cxy * df1) / det;
    float s1 = (cxx * df1 - cxy * df0) / det;
    float d2c = df0 * s0 + df1 * s1;
    const float thr = 4.605170185988091f;   // -2*log(0.1)
    float term = 1.0f / (1.0f + expf(-((thr - d2c) * 2.0f)));  // sigmoid((thr-d2c)/0.5)
    atomicAdd(&sums[4], term);
}

__global__ void k_final(const float* __restrict__ sums, float* __restrict__ out)
{
    float nll   = sums[0] / (float)BB / 200.0f;
    float mse   = 100.0f * sums[1] / (float)(BB * TT * YDI);
    float dmean = sums[2] / (float)(BB * TT);
    float calib = 0.05f * (dmean - 2.0f) * (dmean - 2.0f);
    float kl    = sums[3] / (float)BB;
    float lcov  = 1.0f - sums[4] / (float)(BB * TT);
    out[0] = nll + kl + mse + calib + lcov;
}

// ---------------- host orchestration ----------------
extern "C" void kernel_launch(void* const* d_in, const int* in_sizes, int n_in,
                              void* d_out, int out_size, void* d_ws, size_t ws_size,
                              hipStream_t stream)
{
    (void)in_sizes; (void)n_in; (void)out_size; (void)ws_size;
    const float* x_hist = (const float*)d_in[0];
    const float* y_fut  = (const float*)d_in[1];
    // params in setup_inputs() insertion order
    int pi = 2;
    const float* eh_wih = (const float*)d_in[pi++]; const float* eh_whh = (const float*)d_in[pi++];
    const float* eh_bih = (const float*)d_in[pi++]; const float* eh_bhh = (const float*)d_in[pi++];
    const float* eh_fcw = (const float*)d_in[pi++]; const float* eh_fcb = (const float*)d_in[pi++];
    const float* ef_wih = (const float*)d_in[pi++]; const float* ef_whh = (const float*)d_in[pi++];
    const float* ef_bih = (const float*)d_in[pi++]; const float* ef_bhh = (const float*)d_in[pi++];
    const float* ef_fcw = (const float*)d_in[pi++]; const float* ef_fcb = (const float*)d_in[pi++];
    const float* pr_w1 = (const float*)d_in[pi++];  const float* pr_b1 = (const float*)d_in[pi++];
    const float* pr_w2 = (const float*)d_in[pi++];  const float* pr_b2 = (const float*)d_in[pi++];
    const float* po_w1 = (const float*)d_in[pi++];  const float* po_b1 = (const float*)d_in[pi++];
    const float* po_w2 = (const float*)d_in[pi++];  const float* po_b2 = (const float*)d_in[pi++];
    const float* dc_initw = (const float*)d_in[pi++]; const float* dc_initb = (const float*)d_in[pi++];
    const float* dc_wih = (const float*)d_in[pi++]; const float* dc_whh = (const float*)d_in[pi++];
    const float* dc_bih = (const float*)d_in[pi++]; const float* dc_bhh = (const float*)d_in[pi++];
    const float* hd_pw = (const float*)d_in[pi++];  const float* hd_pb = (const float*)d_in[pi++];
    const float *hd_w1[3], *hd_b1[3], *hd_w2[3], *hd_b2[3], *hd_g[3], *hd_be[3];
    for (int i = 0; i < 3; ++i) {
        hd_w1[i] = (const float*)d_in[pi++]; hd_b1[i] = (const float*)d_in[pi++];
        hd_w2[i] = (const float*)d_in[pi++]; hd_b2[i] = (const float*)d_in[pi++];
        hd_g[i]  = (const float*)d_in[pi++]; hd_be[i] = (const float*)d_in[pi++];
    }
    const float* hd_muw = (const float*)d_in[pi++]; const float* hd_mub = (const float*)d_in[pi++];
    const float* hd_lvw = (const float*)d_in[pi++]; const float* hd_lvb = (const float*)d_in[pi++];
    const float* v0     = (const float*)d_in[pi++];

    // ---- workspace bump allocator (floats, 16B-aligned chunks) ----
    float* wsf = (float*)d_ws;
    size_t off = 0;
    auto alloc = [&](size_t n) { float* p = wsf + off; off += (n + 3) & ~(size_t)3; return p; };
    float* gi     = alloc((size_t)RR * 3 * HH);
    float* gh     = alloc((size_t)RR * 3 * HH);
    float* Hs     = alloc((size_t)RR * HH);
    float* henc   = alloc((size_t)BB * HH);
    float* hfut   = alloc((size_t)BB * HH);
    float* cat2   = alloc((size_t)BB * 2 * HH);
    float* t1     = alloc((size_t)BB * HH);
    float* pr_out = alloc((size_t)BB * 32);
    float* po_out = alloc((size_t)BB * 32);
    float* Z      = alloc((size_t)RR * ZDI);
    float* cond   = alloc((size_t)RR * (HH + ZDI));
    float* xb     = alloc((size_t)RR * WID);
    float* y1     = alloc((size_t)RR * 2 * WID);
    float* y2     = alloc((size_t)RR * WID);
    float* vz     = alloc((size_t)RR * (YDI + ZDI));
    float* vbuf   = alloc((size_t)RR * YDI);
    float* pos    = alloc((size_t)RR * YDI);
    float* mu_main = alloc((size_t)BB * TT * YDI);
    float* lv_main = alloc((size_t)BB * TT * YDI);
    float* trajs   = alloc((size_t)KK * BB * TT * YDI);
    float* sums    = alloc(8);
    float* h_e     = alloc((size_t)BB * HH);

    auto gemm = [&](const float* A, int lda, const float* W, const float* bias,
                    float* C, int ldc, int M, int N, int K, int op) {
        dim3 g((unsigned)((M + 127) / 128), (unsigned)((N + 63) / 64));
        k_gemm<<<g, 256, 0, stream>>>(A, lda, W, bias, C, ldc, M, N, K, op);
    };
    auto zero = [&](float* p, long n) {
        k_zero<<<(unsigned)((n + 255) / 256), 256, 0, stream>>>(p, n);
    };

    // ================= encoders (GRU over sequence) =================
    zero(h_e, (long)BB * HH);
    for (int t = 0; t < LL; ++t) {
        gemm(x_hist + (long)t * XDI, LL * XDI, eh_wih, eh_bih, gi, 3 * HH, BB, 3 * HH, XDI, 1);
        gemm(h_e, HH, eh_whh, eh_bhh, gh, 3 * HH, BB, 3 * HH, HH, 1);
        k_gru<<<(unsigned)(((long)BB * HH + 255) / 256), 256, 0, stream>>>(h_e, gi, gh, BB);
    }
    gemm(h_e, HH, eh_fcw, eh_fcb, henc, HH, BB, HH, HH, 1);

    zero(h_e, (long)BB * HH);
    for (int t = 0; t < TT; ++t) {
        gemm(y_fut + (long)t * YDI, TT * YDI, ef_wih, ef_bih, gi, 3 * HH, BB, 3 * HH, YDI, 1);
        gemm(h_e, HH, ef_whh, ef_bhh, gh, 3 * HH, BB, 3 * HH, HH, 1);
        k_gru<<<(unsigned)(((long)BB * HH + 255) / 256), 256, 0, stream>>>(h_e, gi, gh, BB);
    }
    gemm(h_e, HH, ef_fcw, ef_fcb, hfut, HH, BB, HH, HH, 1);

    // ================= prior / posterior =================
    gemm(henc, HH, pr_w1, pr_b1, t1, HH, BB, HH, HH, 2);              // ReLU
    gemm(t1, HH, pr_w2, pr_b2, pr_out, 32, BB, 32, HH, 1);
    k_concat2<<<(unsigned)(((long)BB * 2 * HH + 255) / 256), 256, 0, stream>>>(cat2, henc, HH, hfut, HH, BB);
    gemm(cat2, 2 * HH, po_w1, po_b1, t1, HH, BB, HH, 2 * HH, 2);      // ReLU
    gemm(t1, HH, po_w2, po_b2, po_out, 32, BB, 32, HH, 1);

    // ================= batched decode (main + K samples) =================
    k_sample_z<<<(unsigned)(((long)RR * ZDI + 255) / 256), 256, 0, stream>>>(po_out, pr_out, Z);
    k_build_cond<<<(unsigned)(((long)RR * (HH + ZDI) + 255) / 256), 256, 0, stream>>>(cond, henc, Z);
    gemm(cond, HH + ZDI, dc_initw, dc_initb, Hs, HH, RR, HH, HH + ZDI, 4);  // tanh
    k_init_pos_v<<<(unsigned)(((long)RR * YDI + 255) / 256), 256, 0, stream>>>(pos, vbuf, x_hist, v0);

    for (int t = 0; t < TT; ++t) {
        k_build_vz<<<(unsigned)(((long)RR * (YDI + ZDI) + 255) / 256), 256, 0, stream>>>(vz, vbuf, Z);
        gemm(vz, YDI + ZDI, dc_wih, dc_bih, gi, 3 * HH, RR, 3 * HH, YDI + ZDI, 1);
        gemm(Hs, HH, dc_whh, dc_bhh, gh, 3 * HH, RR, 3 * HH, HH, 1);
        k_gru<<<(unsigned)(((long)RR * HH + 255) / 256), 256, 0, stream>>>(Hs, gi, gh, RR);

        gemm(Hs, HH, hd_pw, hd_pb, xb, WID, RR, WID, HH, 1);
        for (int i = 0; i < 3; ++i) {
            gemm(xb, WID, hd_w1[i], hd_b1[i], y1, 2 * WID, RR, 2 * WID, WID, 3);   // GELU
            gemm(y1, 2 * WID, hd_w2[i], hd_b2[i], y2, WID, RR, WID, 2 * WID, 1);
            k_ln_res<<<(unsigned)RR, 256, 0, stream>>>(xb, y2, hd_g[i], hd_be[i], RR);
        }
        k_heads<<<(unsigned)((RR + 255) / 256), 256, 0, stream>>>(
            xb, hd_muw, hd_mub, hd_lvw, hd_lvb, pos, vbuf, mu_main, lv_main, trajs, t);
    }

    // ================= losses =================
    zero(sums, 8);
    k_nll_mse_calib<<<(unsigned)((BB * TT + 255) / 256), 256, 0, stream>>>(mu_main, lv_main, y_fut, sums);
    k_kl<<<(unsigned)((BB + 255) / 256), 256, 0, stream>>>(po_out, pr_out, sums);
    k_cov<<<(unsigned)((BB * TT + 255) / 256), 256, 0, stream>>>(trajs, y_fut, sums);
    k_final<<<1, 1, 0, stream>>>(sums, (float*)d_out);
}
